// TimeMixing_54735063220577
// MI455X (gfx1250) — matrix operand order repaired and checked by reading, not compiled, not run on hardware
//
#include <hip/hip_runtime.h>
#include <hip/hip_bf16.h>

// ---------------------------------------------------------------------------
// RWKV-style TimeMixing for MI455X (gfx1250, wave32).
//   B=8, N=4096, D=1024, H=16, HD=64.
// Pipeline:
//   1) prep_weights : f32 [D,D] row-major -> fragment-major bf16 (WMMA B-operand
//      layout, 32B contiguous per lane) for Wg,Wr,Wk,Wv,Wo.
//   2) prep_decay   : decay = exp(-exp(w)).
//   3) gemm<1> (z=0..3): g/r/k/v = (m*x+(1-m)*roll(x)) @ W^T + b via
//      v_wmma_f32_16x16x32_bf16, f32 accum. Mix fused into the A-fragment
//      build (compile-time specialized, float4 loads, no branches).
//   4) chunked decay scan (3 passes, 32 chunks x 128): local suffix sums,
//      carry recurrence with dec^L, replay fused with LayerNorm(64) +
//      sigmoid(g) gate; y written in place over g.
//   5) gemm<0>: out = y @ Wo^T + b -> d_out.
// HBM-bound (~2.3 GB moved vs 344 GFLOP of bf16 WMMA): the GEMM A/B streams
// must be pure b128 traffic, which is what the DO_MIX template guarantees.
// ---------------------------------------------------------------------------

#define B_BATCH 8
#define N_SEQ   4096
#define D_DIM   1024
#define H_HEADS 16
#define HD_DIM  64
#define CHUNKS  32
#define CHUNK_L (N_SEQ / CHUNKS)   // 128
#define KTILES  (D_DIM / 32)       // 32
#define NTILES  (D_DIM / 16)       // 64

typedef __bf16 v16bf __attribute__((ext_vector_type(16)));
typedef float  v8f   __attribute__((ext_vector_type(8)));

// -------------------- 1) weight fragment pre-swizzle -----------------------
// B-operand layout for V_WMMA_*_16X16X32_BF16 (32x16, K x N):
//   lane l: n = l%16, K = kt*32 + ((l<16)?0:8) + (i<8 ? i : i+8), i = 0..15.
// Fragment index: (((w*KTILES + kt)*NTILES + nt)*32 + lane)*16 + i
__global__ __launch_bounds__(256) void prep_weights_kernel(
    const float* __restrict__ Wg, const float* __restrict__ Wr,
    const float* __restrict__ Wk, const float* __restrict__ Wv,
    const float* __restrict__ Wo, __bf16* __restrict__ Wfrag)
{
    int gid  = blockIdx.x * blockDim.x + threadIdx.x;   // 5*KTILES*NTILES*32
    int lane = gid & 31;
    int t    = gid >> 5;
    int nt   = t & (NTILES - 1); t >>= 6;
    int kt   = t & (KTILES - 1); t >>= 5;
    int w    = t;
    if (w >= 5) return;
    const float* W = (w == 0) ? Wg : (w == 1) ? Wr : (w == 2) ? Wk
                   : (w == 3) ? Wv : Wo;
    int ncol  = nt * 16 + (lane & 15);
    int kbase = kt * 32 + ((lane < 16) ? 0 : 8);
    __bf16* dst = Wfrag +
        ((((size_t)w * KTILES + kt) * NTILES + nt) * 32 + (size_t)lane) * 16;
    const float* src = W + (size_t)ncol * D_DIM + kbase;
#pragma unroll
    for (int i = 0; i < 16; ++i) dst[i] = (__bf16)src[(i < 8) ? i : i + 8];
}

// -------------------- 2) decay = exp(-exp(w)) ------------------------------
__global__ void prep_decay_kernel(const float* __restrict__ w,
                                  float* __restrict__ decay)
{
    int i = blockIdx.x * blockDim.x + threadIdx.x;
    if (i < H_HEADS * HD_DIM) decay[i] = __expf(-__expf(w[i]));
}

// -------------------- A-fragment build (mix fused, float4 loads) -----------
// A-fragment (16x32 bf16) is thread-private per the ISA layout:
//   lane<16:  elems 0..7 -> K k0+0..7,  elems 8..15 -> K k0+16..23
//   lane>=16: elems 0..7 -> K k0+8..15, elems 8..15 -> K k0+24..31
template <int DO_MIX>
__device__ __forceinline__ v16bf make_a_frag(
    const float* __restrict__ xrow, const float* __restrict__ xprev,
    const float* __restrict__ mvec, int k0, int kb)
{
    float xv[16], pv[16], mm[16];
    const float4* x0 = (const float4*)(xrow + k0 + kb);
    const float4* x1 = (const float4*)(xrow + k0 + 16 + kb);
    *(float4*)&xv[0]  = x0[0];  *(float4*)&xv[4]  = x0[1];
    *(float4*)&xv[8]  = x1[0];  *(float4*)&xv[12] = x1[1];
    if (DO_MIX) {
        const float4* p0 = (const float4*)(xprev + k0 + kb);
        const float4* p1 = (const float4*)(xprev + k0 + 16 + kb);
        const float4* m0 = (const float4*)(mvec + k0 + kb);
        const float4* m1 = (const float4*)(mvec + k0 + 16 + kb);
        *(float4*)&pv[0]  = p0[0];  *(float4*)&pv[4]  = p0[1];
        *(float4*)&pv[8]  = p1[0];  *(float4*)&pv[12] = p1[1];
        *(float4*)&mm[0]  = m0[0];  *(float4*)&mm[4]  = m0[1];
        *(float4*)&mm[8]  = m1[0];  *(float4*)&mm[12] = m1[1];
    }
    v16bf a;
#pragma unroll
    for (int i = 0; i < 16; ++i) {
        float v = xv[i];
        if (DO_MIX) v = mm[i] * v + (1.f - mm[i]) * pv[i];
        a[i] = (__bf16)v;
    }
    return a;
}

// -------------------- 3/5) WMMA GEMM ---------------------------------------
// Block: 256 threads = 8 waves. Block tile: 256 (rows) x 64 (cols).
// Each wave: 32 rows x 64 cols = 2 A-fragments x 4 B-fragments = 8 WMMA
// accumulators per k-step; every B fragment feeds two WMMAs.
template <int DO_MIX>
__global__ __launch_bounds__(256) void gemm_kernel(
    const float*  __restrict__ X,      // [B*N, D] (x, or y for the Wo pass)
    const __bf16* __restrict__ Wfrag,  // fragment-major weights, D*D per z
    const float* b0, const float* b1, const float* b2, const float* b3,
    const float* m0, const float* m1, const float* m2, const float* m3,
    float* Out)                        // base; +z*B*N*D slice
{
    const int lane  = threadIdx.x & 31;
    const int wave  = threadIdx.x >> 5;
    const int z     = blockIdx.z;
    const int ncol0 = blockIdx.x * 64;
    const int row0  = blockIdx.y * 256;

    const float* bias = (z == 0) ? b0 : (z == 1) ? b1 : (z == 2) ? b2 : b3;
    const float* mvec = (z == 0) ? m0 : (z == 1) ? m1 : (z == 2) ? m2 : m3;
    const __bf16* wbase = Wfrag + (size_t)z * D_DIM * D_DIM;

    // the two rows this lane's A fragments cover (token-shift wraps in-batch)
    const float* xrow[2];
    const float* xprev[2];
#pragma unroll
    for (int t = 0; t < 2; ++t) {
        const int grow  = row0 + wave * 32 + t * 16 + (lane & 15);
        const int b     = grow >> 12;                 // / N_SEQ
        const int n     = grow & (N_SEQ - 1);
        const int nprev = (n == 0) ? (N_SEQ - 1) : (n - 1);
        xrow[t]  = X + (size_t)(b * N_SEQ + n)     * D_DIM;
        xprev[t] = X + (size_t)(b * N_SEQ + nprev) * D_DIM;
    }
    const int kb = (lane < 16) ? 0 : 8;

    v8f acc[2][4];
#pragma unroll
    for (int t = 0; t < 2; ++t)
#pragma unroll
        for (int s = 0; s < 4; ++s)
            acc[t][s] = (v8f){0.f,0.f,0.f,0.f,0.f,0.f,0.f,0.f};

    for (int kt = 0; kt < KTILES; ++kt) {
        const int k0 = kt * 32;
        v16bf a0 = make_a_frag<DO_MIX>(xrow[0], xprev[0], mvec, k0, kb);
        v16bf a1 = make_a_frag<DO_MIX>(xrow[1], xprev[1], mvec, k0, kb);
        if (kt + 1 < KTILES) {  // prefetch next k-tile of the B stream
            __builtin_prefetch(wbase +
                (((size_t)(kt + 1) * NTILES + (ncol0 >> 4)) * 32 + lane) * 16,
                0, 1);
        }
#pragma unroll
        for (int s = 0; s < 4; ++s) {
            const int nt = (ncol0 >> 4) + s;
            const v16bf* bp = (const v16bf*)(wbase +
                (((size_t)kt * NTILES + nt) * 32 + (size_t)lane) * 16);
            v16bf bf = *bp;   // one contiguous 32B load per lane
            acc[0][s] = __builtin_amdgcn_wmma_f32_16x16x32_bf16(
                false, a0, false, bf, (short)0, acc[0][s], false, false);
            acc[1][s] = __builtin_amdgcn_wmma_f32_16x16x32_bf16(
                false, a1, false, bf, (short)0, acc[1][s], false, false);
        }
    }

    // Epilogue: C/D layout -> M = r + 8*(lane>=16), N = lane%16.
    float* outp = Out + (size_t)z * ((size_t)B_BATCH * N_SEQ * D_DIM);
    const int mhi = (lane >> 4) * 8;
#pragma unroll
    for (int t = 0; t < 2; ++t) {
#pragma unroll
        for (int s = 0; s < 4; ++s) {
            const int c    = ncol0 + s * 16 + (lane & 15);
            const float bv = bias[c];
#pragma unroll
            for (int r = 0; r < 8; ++r) {
                const int rr = row0 + wave * 32 + t * 16 + mhi + r;
                outp[(size_t)rr * D_DIM + c] = acc[t][s][r] + bv;
            }
        }
    }
}

// -------------------- 4a) chunk-local suffix sums --------------------------
__global__ __launch_bounds__(64) void scan_chunk_sum_kernel(
    const float* __restrict__ K, const float* __restrict__ V,
    const float* __restrict__ decay, float* __restrict__ S)
{
    const int blk = blockIdx.x;            // ((b*H + h)*CHUNKS + c)
    const int c   = blk & (CHUNKS - 1);
    const int bh  = blk >> 5;
    const int h   = bh & (H_HEADS - 1);
    const int b   = bh >> 4;
    const int hd  = threadIdx.x;
    const float dec = decay[h * HD_DIM + hd];
    size_t base = (size_t)(b * N_SEQ + c * CHUNK_L) * D_DIM + h * HD_DIM + hd;
    float s = 0.f;
    for (int i = 0; i < CHUNK_L; ++i) {
        size_t idx = base + (size_t)i * D_DIM;
        s = dec * s + K[idx] * V[idx];
    }
    S[(size_t)blk * HD_DIM + hd] = s;
}

// -------------------- 4b) carry recurrence across chunks -------------------
// carry_c = A_{c-1};  A_c = dec^L * A_{c-1} + S_c.  S overwritten with carries.
__global__ void scan_carry_kernel(const float* __restrict__ decay, float* S)
{
    int gid = blockIdx.x * blockDim.x + threadIdx.x;   // B*H*HD = 8192
    if (gid >= B_BATCH * H_HEADS * HD_DIM) return;
    const int hd = gid & (HD_DIM - 1);
    const int h  = (gid >> 6) & (H_HEADS - 1);
    const int b  = gid >> 10;
    const float dec  = decay[h * HD_DIM + hd];
    const float decL = powf(dec, (float)CHUNK_L);
    float A = 0.f;
    size_t base = ((size_t)(b * H_HEADS + h) * CHUNKS) * HD_DIM + hd;
    for (int c = 0; c < CHUNKS; ++c) {
        size_t idx = base + (size_t)c * HD_DIM;
        float s = S[idx];
        S[idx]  = A;            // carry into chunk c
        A = decL * A + s;
    }
}

// -------------------- 4c) replay + LayerNorm + sigmoid gate ----------------
// Block = 64 threads = one (b,h,chunk). y = sigmoid(g)*LN(r*wkv), in place
// over G. LN over HD=64 via wave32 shuffles + 2-wave LDS combine.
__global__ __launch_bounds__(64) void scan_ln_gate_kernel(
    float* G, const float* __restrict__ R,
    const float* __restrict__ K, const float* __restrict__ V,
    const float* __restrict__ carry, const float* __restrict__ decay,
    const float* __restrict__ u)
{
    const int blk = blockIdx.x;
    const int c   = blk & (CHUNKS - 1);
    const int bh  = blk >> 5;
    const int h   = bh & (H_HEADS - 1);
    const int b   = bh >> 4;
    const int hd  = threadIdx.x;
    const int wv  = hd >> 5;
    const float dec = decay[h * HD_DIM + hd];
    const float uu  = u[h * HD_DIM + hd];
    float acc = carry[(size_t)blk * HD_DIM + hd];
    __shared__ float s_sum[2], s_sq[2];
    size_t base = (size_t)(b * N_SEQ + c * CHUNK_L) * D_DIM + h * HD_DIM + hd;
    for (int i = 0; i < CHUNK_L; ++i) {
        size_t idx = base + (size_t)i * D_DIM;
        float kv  = K[idx] * V[idx];
        acc       = dec * acc + kv;
        float wkv = acc + uu * kv;
        float rw  = R[idx] * wkv;
        float s = rw, q = rw * rw;
#pragma unroll
        for (int off = 16; off > 0; off >>= 1) {
            s += __shfl_xor(s, off, 32);
            q += __shfl_xor(q, off, 32);
        }
        if ((hd & 31) == 0) { s_sum[wv] = s; s_sq[wv] = q; }
        __syncthreads();
        float mean = (s_sum[0] + s_sum[1]) * (1.f / HD_DIM);
        float var  = (s_sq[0]  + s_sq[1])  * (1.f / HD_DIM) - mean * mean;
        __syncthreads();
        float ln = (rw - mean) * rsqrtf(var + 1e-5f);
        float g  = G[idx];
        G[idx]   = ln / (1.f + __expf(-g));   // y, in place
    }
}

// ---------------------------------------------------------------------------
extern "C" void kernel_launch(void* const* d_in, const int* in_sizes, int n_in,
                              void* d_out, int out_size, void* d_ws, size_t ws_size,
                              hipStream_t stream)
{
    (void)in_sizes; (void)n_in; (void)out_size; (void)ws_size;
    const float* x    = (const float*)d_in[0];
    const float* Wg_w = (const float*)d_in[1];
    const float* Wg_b = (const float*)d_in[2];
    const float* Wr_w = (const float*)d_in[3];
    const float* Wr_b = (const float*)d_in[4];
    const float* Wk_w = (const float*)d_in[5];
    const float* Wk_b = (const float*)d_in[6];
    const float* Wv_w = (const float*)d_in[7];
    const float* Wv_b = (const float*)d_in[8];
    const float* Wo_w = (const float*)d_in[9];
    const float* Wo_b = (const float*)d_in[10];
    const float* mg   = (const float*)d_in[11];
    const float* mr   = (const float*)d_in[12];
    const float* mk   = (const float*)d_in[13];
    const float* mv   = (const float*)d_in[14];
    const float* w    = (const float*)d_in[15];
    const float* u    = (const float*)d_in[16];
    float* out = (float*)d_out;

    const size_t MD = (size_t)B_BATCH * N_SEQ * D_DIM;   // 33,554,432 elems
    char* ws = (char*)d_ws;
    size_t off = 0;
    auto align256 = [](size_t v) { return (v + 255) & ~(size_t)255; };

    __bf16* Wfrag = (__bf16*)(ws + off);
    off = align256(off + (size_t)5 * D_DIM * D_DIM * sizeof(__bf16)); // 10 MB
    float* decay = (float*)(ws + off);
    off = align256(off + (size_t)H_HEADS * HD_DIM * sizeof(float));
    float* Sbuf = (float*)(ws + off);
    off = align256(off + (size_t)B_BATCH * H_HEADS * CHUNKS * HD_DIM * sizeof(float));
    float* GRKV = (float*)(ws + off);                    // 4 x 128 MB
    float* Gb = GRKV;
    float* Rb = GRKV + MD;
    float* Kb = GRKV + 2 * MD;
    float* Vb = GRKV + 3 * MD;

    // 1) weight fragments (5 * KTILES * NTILES * 32 lanes = 327,680 threads)
    prep_weights_kernel<<<dim3(1280), dim3(256), 0, stream>>>(
        Wg_w, Wr_w, Wk_w, Wv_w, Wo_w, Wfrag);

    // 2) decay
    prep_decay_kernel<<<dim3(4), dim3(256), 0, stream>>>(w, decay);

    // 3) g,r,k,v projections with fused token-shift mix
    gemm_kernel<1><<<dim3(D_DIM / 64, (B_BATCH * N_SEQ) / 256, 4),
                     dim3(256), 0, stream>>>(
        x, Wfrag, Wg_b, Wr_b, Wk_b, Wv_b, mg, mr, mk, mv, GRKV);

    // 4) chunked decay scan
    scan_chunk_sum_kernel<<<dim3(B_BATCH * H_HEADS * CHUNKS), dim3(HD_DIM),
                            0, stream>>>(Kb, Vb, decay, Sbuf);
    scan_carry_kernel<<<dim3(32), dim3(256), 0, stream>>>(decay, Sbuf);
    scan_ln_gate_kernel<<<dim3(B_BATCH * H_HEADS * CHUNKS), dim3(HD_DIM),
                          0, stream>>>(Gb, Rb, Kb, Vb, Sbuf, decay, u);

    // 5) output projection: out = y @ Wo^T + b  (y lives in Gb, no mixing)
    gemm_kernel<0><<<dim3(D_DIM / 64, (B_BATCH * N_SEQ) / 256, 1),
                     dim3(256), 0, stream>>>(
        Gb, Wfrag + (size_t)4 * D_DIM * D_DIM, Wo_b, nullptr, nullptr, nullptr,
        nullptr, nullptr, nullptr, nullptr, out);
}
